// AttentionLayer_16939351015520
// MI455X (gfx1250) — compile-verified, matrix operand order checked by reading, not run on hardware
//
#include <hip/hip_runtime.h>

// ---------------------------------------------------------------------------
// MI455X (gfx1250) fused attention layer.  B=4, N=2048, D=1024, H=16, DK=64.
// ~137 GFLOP, <200MB essential traffic -> matrix-core bound.
// All heavy math: v_wmma_f32_16x16x32_bf16.
// Pipeline (all on `stream`):
//   1) X: f32 -> bf16 (vectorized RNE). Weights: f32 -> bf16 + transpose
//      (both GEMM operands K-contiguous -> async b128 LDS copies).
//   2) WMMA GEMM (double-buffered GLOBAL_LOAD_ASYNC_TO_LDS_B128 pipeline):
//        X@Wq -> Q*0.125 [B,H,N,DK] (1/sqrt(DK) folded in, exact: 2^-3),
//        X@Wk -> K [B,H,N,DK], X@Wv -> V^T [B,H,DK,N]
//   3) flash attention (online softmax; diagonal kv block peeled so the
//      causal-mask VALU never runs in the O(N^2) main loop) -> Ocat bf16
//   4) WMMA GEMM Ocat@Wo -> d_out f32
// ---------------------------------------------------------------------------

typedef unsigned short u16s;
typedef __attribute__((ext_vector_type(16))) __bf16       v16bf;
typedef __attribute__((ext_vector_type(8)))  float        v8f;
typedef __attribute__((ext_vector_type(4)))  float        v4f;
typedef __attribute__((ext_vector_type(4)))  unsigned int v4u;
typedef __attribute__((ext_vector_type(2)))  unsigned int v2u;

#define DEV __device__ __forceinline__

static constexpr int Bsz  = 4;
static constexpr int SeqN = 2048;
static constexpr int Dm   = 1024;
static constexpr int Hh   = 16;
static constexpr int Dk   = 64;
static constexpr int Mrows = Bsz * SeqN;      // 8192

DEV u16s bf16_rne(float x) {
  unsigned int u = __float_as_uint(x);
  u += 0x7FFFu + ((u >> 16) & 1u);            // round-to-nearest-even
  return (u16s)(u >> 16);
}

// A/B operand fragment: 16 bf16 across 8 VGPRs, filled by two b128 loads.
// ISA 16-bit A 16x32 layout: lane owns row M=lane&15; v0..3 = K kh*8..+7,
// v4..7 = K 16+kh*8..+7 (kh = lane>>4). B mirrors with lane = out column,
// so B sources must be K-contiguous per column.
union FragAB { v16bf bf; v4u q[2]; };
union FragC  { v8f v; float f[8]; };

DEV v8f wmma_bf16(v16bf a, v16bf b, v8f c) {
  return __builtin_amdgcn_wmma_f32_16x16x32_bf16(
      false, a, false, b, (short)0, c, false, false);
}

DEV FragAB load_frag(const u16s* p) {   // p = row base + chunk*32 + kh*8
  FragAB f;
  f.q[0] = *(const v4u*)(p);
  f.q[1] = *(const v4u*)(p + 16);
  return f;
}

// Async global->LDS DMA, 16 bytes per lane, tracked by ASYNCcnt.
// LDS byte address = low 32 bits of the generic shared pointer.
DEV void async_copy_b128(const u16s* gsrc, u16s* ldst) {
  unsigned lds = (unsigned)(size_t)(void*)ldst;
  unsigned long long ga = (unsigned long long)(size_t)gsrc;
  asm volatile("global_load_async_to_lds_b128 %0, %1, off"
               :: "v"(lds), "v"(ga) : "memory");
}
DEV void wait_async0() { asm volatile("s_wait_asynccnt 0x0" ::: "memory"); }

// ---------------------------------------------------------------------------
// Kernel 1a: f32 -> bf16, 4 elements/thread (memory bound)
// ---------------------------------------------------------------------------
__global__ __launch_bounds__(256) void cvt_f32_bf16_x4(
    const float* __restrict__ s, u16s* __restrict__ d, int n4) {
  for (int i = blockIdx.x * blockDim.x + threadIdx.x; i < n4;
       i += gridDim.x * blockDim.x) {
    v4f x = *(const v4f*)(s + 4 * (size_t)i);
    v2u o;
    o.x = ((unsigned)bf16_rne(x.y) << 16) | bf16_rne(x.x);
    o.y = ((unsigned)bf16_rne(x.w) << 16) | bf16_rne(x.z);
    *(v2u*)(d + 4 * (size_t)i) = o;
  }
}

// ---------------------------------------------------------------------------
// Kernel 1b: W[K,N] f32 -> W^T[N,K] bf16 (32x32 LDS tile transpose)
// ---------------------------------------------------------------------------
__global__ __launch_bounds__(256) void cvt_transpose_bf16(
    const float* __restrict__ W, u16s* __restrict__ Wt, int Kd, int Nd) {
  __shared__ u16s tile[32][33];
  const int bn = blockIdx.x * 32, bk = blockIdx.y * 32;
  const int c = threadIdx.x & 31, r0 = threadIdx.x >> 5;   // 8 rows/pass
#pragma unroll
  for (int i = 0; i < 4; ++i) {
    int r = r0 + i * 8;
    tile[r][c] = bf16_rne(W[(size_t)(bk + r) * Nd + bn + c]);
  }
  __syncthreads();
#pragma unroll
  for (int i = 0; i < 4; ++i) {
    int r = r0 + i * 8;
    Wt[(size_t)(bn + r) * Kd + bk + c] = tile[c][r];
  }
}

// ---------------------------------------------------------------------------
// Kernel 2: bf16 WMMA GEMM, C[M,Nn] = escale * A[M,K] * Bt[Nn,K]^T
//   128x128 tile, BK=32, 256 threads = 8 waves (4 M x 2 N), 8 WMMAs/step.
//   LDS tiles filled by GLOBAL_LOAD_ASYNC_TO_LDS_B128, double buffered so
//   the DMA of tile kt+32 overlaps the WMMAs of tile kt.
// EPI 0: bf16 -> [B,H,N,DK]   EPI 1: bf16 -> [B,H,DK,N]   EPI 2: f32 [M,Nn]
// ---------------------------------------------------------------------------
template <int EPI>
__global__ __launch_bounds__(256) void gemm_bf16(
    const u16s* __restrict__ A, const u16s* __restrict__ Bt,
    void* __restrict__ out, int M, int K, int Nn, float escale) {
  __shared__ alignas(16) u16s ls[2][2][128 * 40];  // [buf][A/B][row][32k+pad]

  const int tid  = threadIdx.x;
  const int lane = tid & 31;
  const int wid  = tid >> 5;
  const int l15  = lane & 15;
  const int kh   = lane >> 4;
  const int wm   = wid & 3;                    // 32-row strip
  const int wn   = wid >> 2;                   // 64-col strip
  const int blockN = blockIdx.x * 128;
  const int blockM = blockIdx.y * 128;

  // Each thread DMAs 2 x 16B for A and for B per K-step (512 transfers/tile)
  auto issue_tile = [&](int kt, int buf) {
#pragma unroll
    for (int i = 0; i < 2; ++i) {
      int idx = tid + i * 256;
      int r = idx >> 2, cch = idx & 3;           // row 0..127, 8-elem chunk
      async_copy_b128(A  + (size_t)(blockM + r) * K + kt + cch * 8,
                      &ls[buf][0][r * 40 + cch * 8]);
      async_copy_b128(Bt + (size_t)(blockN + r) * K + kt + cch * 8,
                      &ls[buf][1][r * 40 + cch * 8]);
    }
  };

  FragC acc[2][4] = {};

  issue_tile(0, 0);
  wait_async0();
  __syncthreads();

  int cur = 0;
  for (int kt = 0; kt < K; kt += 32) {
    const int nxt = cur ^ 1;
    if (kt + 32 < K) issue_tile(kt + 32, nxt);   // overlap DMA with WMMA

    FragAB af[2];
#pragma unroll
    for (int fm = 0; fm < 2; ++fm)
      af[fm] = load_frag(&ls[cur][0][(wm * 32 + fm * 16 + l15) * 40 + kh * 8]);
#pragma unroll
    for (int fn = 0; fn < 4; ++fn) {
      FragAB bfr =
          load_frag(&ls[cur][1][(wn * 64 + fn * 16 + l15) * 40 + kh * 8]);
#pragma unroll
      for (int fm = 0; fm < 2; ++fm)
        acc[fm][fn].v = wmma_bf16(af[fm].bf, bfr.bf, acc[fm][fn].v);
    }

    wait_async0();        // own DMAs into nxt landed
    __syncthreads();      // everyone done reading cur + their DMAs landed
    cur = nxt;
  }

  // Epilogue. C layout: m = v + 8*kh, n = l15 within each 16x16 frag.
#pragma unroll
  for (int fm = 0; fm < 2; ++fm)
#pragma unroll
    for (int fn = 0; fn < 4; ++fn)
#pragma unroll
      for (int v = 0; v < 8; ++v) {
        int row = blockM + wm * 32 + fm * 16 + kh * 8 + v;
        int col = blockN + wn * 64 + fn * 16 + l15;
        float val = acc[fm][fn].f[v] * escale;
        if constexpr (EPI == 2) {
          ((float*)out)[(size_t)row * Nn + col] = val;
        } else {
          int bb = row >> 11, ns = row & (SeqN - 1);
          int hh = col >> 6,  dk = col & (Dk - 1);
          size_t bh = (size_t)bb * Hh + hh;
          if constexpr (EPI == 0)
            ((u16s*)out)[(bh * SeqN + ns) * Dk + dk] = bf16_rne(val);
          else
            ((u16s*)out)[(bh * Dk + dk) * SeqN + ns] = bf16_rne(val);
        }
      }
}

// ---------------------------------------------------------------------------
// Kernel 3: flash attention (causal), bf16 WMMA, online softmax.
//   grid = (N/64, B*H); 128 threads = 4 waves; wave w owns rows qi*64+w*16..
//   Per kv block of 64: 8 WMMAs for QK^T + 8 WMMAs for P*V.
//   The 1/sqrt(DK) scale is pre-folded into Q; the causal mask exists only
//   in the DIAG=true instantiation (called once), so the main loop carries
//   no compare/select VALU.
// ---------------------------------------------------------------------------
template <bool DIAG>
DEV void kv_step(int kvb, int qrow0, int l15, int kh,
                 const u16s* __restrict__ Kh, const u16s* __restrict__ Vh,
                 u16s* __restrict__ pw, const FragAB* qf, FragC* oacc,
                 float* mrow, float* lrow) {
  // ---- S = Q * K^T  (B operand: lane = kv column -> K rows, b128 loads)
  FragC sacc[4] = {};
#pragma unroll
  for (int fn = 0; fn < 4; ++fn) {
    const u16s* kp = Kh + (size_t)(kvb + fn * 16 + l15) * Dk + kh * 8;
#pragma unroll
    for (int c = 0; c < 2; ++c) {
      FragAB kf = load_frag(kp + c * 32);
      sacc[fn].v = wmma_bf16(qf[c].bf, kf.bf, sacc[fn].v);
    }
  }
  // ---- online softmax (a C-frag row lives in one 16-lane half)
#pragma unroll
  for (int v = 0; v < 8; ++v) {
    int row = qrow0 + kh * 8 + v;
    float rmax = -3.0e38f;
#pragma unroll
    for (int fn = 0; fn < 4; ++fn) {
      float s = sacc[fn].f[v];
      if constexpr (DIAG) {
        if ((kvb + fn * 16 + l15) > row) s = -1.0e9f;
        sacc[fn].f[v] = s;
      }
      rmax = fmaxf(rmax, s);
    }
    rmax = fmaxf(rmax, __shfl_xor(rmax, 1, 32));
    rmax = fmaxf(rmax, __shfl_xor(rmax, 2, 32));
    rmax = fmaxf(rmax, __shfl_xor(rmax, 4, 32));
    rmax = fmaxf(rmax, __shfl_xor(rmax, 8, 32));
    float mn    = fmaxf(mrow[v], rmax);
    float alpha = __expf(mrow[v] - mn);
    mrow[v] = mn;
    float rs = 0.0f;
#pragma unroll
    for (int fn = 0; fn < 4; ++fn) {
      float p = __expf(sacc[fn].f[v] - mn);
      sacc[fn].f[v] = p;
      rs += p;
    }
    rs += __shfl_xor(rs, 1, 32);
    rs += __shfl_xor(rs, 2, 32);
    rs += __shfl_xor(rs, 4, 32);
    rs += __shfl_xor(rs, 8, 32);
    lrow[v] = lrow[v] * alpha + rs;
#pragma unroll
    for (int fn = 0; fn < 4; ++fn) oacc[fn].f[v] *= alpha;
    // C-layout -> LDS (per-wave region; same-wave LDS ops are in-order)
#pragma unroll
    for (int fn = 0; fn < 4; ++fn)
      pw[(kh * 8 + v) * 72 + fn * 16 + l15] = bf16_rne(sacc[fn].f[v]);
  }
  asm volatile("" ::: "memory");   // keep P loads after P stores

  // ---- reload P in A-fragment layout (two ds_load_b128 per chunk)
  FragAB pf[2];
#pragma unroll
  for (int c = 0; c < 2; ++c)
    pf[c] = load_frag(pw + (size_t)l15 * 72 + c * 32 + kh * 8);

  // ---- O += P * V   (V^T storage -> B operand K-contiguous)
#pragma unroll
  for (int fn = 0; fn < 4; ++fn) {
    const u16s* vp = Vh + (size_t)(fn * 16 + l15) * SeqN + kvb + kh * 8;
#pragma unroll
    for (int c = 0; c < 2; ++c) {
      FragAB vf = load_frag(vp + c * 32);
      oacc[fn].v = wmma_bf16(pf[c].bf, vf.bf, oacc[fn].v);
    }
  }
}

__global__ __launch_bounds__(128) void flash_attn(
    const u16s* __restrict__ Qw,  // [B*H, N, DK] bf16, pre-scaled by 0.125
    const u16s* __restrict__ Kw,  // [B*H, N, DK] bf16
    const u16s* __restrict__ Vt,  // [B*H, DK, N] bf16 (transposed)
    u16s* __restrict__ Ocat) {    // [B, N, H*DK] bf16
  __shared__ alignas(16) u16s pls[4 * 16 * 72];   // per-wave P scratch

  const int tid  = threadIdx.x;
  const int lane = tid & 31;
  const int wid  = tid >> 5;
  const int l15  = lane & 15;
  const int kh   = lane >> 4;
  const int qi   = blockIdx.x;
  const int bh   = blockIdx.y;

  const u16s* Qh = Qw + (size_t)bh * SeqN * Dk;
  const u16s* Kh = Kw + (size_t)bh * SeqN * Dk;
  const u16s* Vh = Vt + (size_t)bh * Dk * SeqN;
  u16s* pw = pls + wid * 16 * 72;

  const int qrow0 = qi * 64 + wid * 16;
  FragAB qf[2];
#pragma unroll
  for (int c = 0; c < 2; ++c)
    qf[c] = load_frag(Qh + (size_t)(qrow0 + l15) * Dk + c * 32 + kh * 8);

  FragC oacc[4] = {};
  float mrow[8], lrow[8];
#pragma unroll
  for (int v = 0; v < 8; ++v) { mrow[v] = -3.0e38f; lrow[v] = 0.0f; }

  for (int j = 0; j < qi; ++j) {     // mask-free main loop
    const int kvb = j * 64;
    __builtin_prefetch(Kh + (size_t)(kvb + 64 + lane) * Dk, 0, 3);
    __builtin_prefetch(Vh + (size_t)lane * SeqN + kvb + 64, 0, 3);
    kv_step<false>(kvb, qrow0, l15, kh, Kh, Vh, pw, qf, oacc, mrow, lrow);
  }
  kv_step<true>(qi * 64, qrow0, l15, kh, Kh, Vh, pw, qf, oacc, mrow, lrow);

  // ---- finalize: O /= l, write bf16 to [B, N, H*DK]
  const int b = bh >> 4, h = bh & 15;
#pragma unroll
  for (int fn = 0; fn < 4; ++fn)
#pragma unroll
    for (int v = 0; v < 8; ++v) {
      int n   = qrow0 + kh * 8 + v;
      int col = h * 64 + fn * 16 + l15;
      float o = oacc[fn].f[v] / lrow[v];
      Ocat[((size_t)b * SeqN + n) * (Hh * Dk) + col] = bf16_rne(o);
    }
}

// ---------------------------------------------------------------------------
// Host launcher. Inputs: X, mask(unused; causal analytic), Wq, Wk, Wv, Wo.
// Workspace (bf16): Xb | WqT,WkT,WvT,WoT | Q | K | V^T | Ocat   (~88MB)
// ---------------------------------------------------------------------------
extern "C" void kernel_launch(void* const* d_in, const int* in_sizes, int n_in,
                              void* d_out, int out_size, void* d_ws,
                              size_t ws_size, hipStream_t stream) {
  const float* X  = (const float*)d_in[0];
  const float* Wq = (const float*)d_in[2];
  const float* Wk = (const float*)d_in[3];
  const float* Wv = (const float*)d_in[4];
  const float* Wo = (const float*)d_in[5];

  const size_t XE = (size_t)Mrows * Dm;        // 8.39M
  const size_t WE = (size_t)Dm * Dm;           // 1.05M
  u16s* Xb  = (u16s*)d_ws;
  u16s* Wqt = Xb  + XE;
  u16s* Wkt = Wqt + WE;
  u16s* Wvt = Wkt + WE;
  u16s* Wot = Wvt + WE;
  u16s* Qb  = Wot + WE;
  u16s* Kb  = Qb  + XE;
  u16s* Vtb = Kb  + XE;
  u16s* Oc  = Vtb + XE;

  // 1) X -> bf16 (vectorized); weights -> bf16 transposed (K-contiguous)
  cvt_f32_bf16_x4<<<2048, 256, 0, stream>>>(X, Xb, (int)(XE / 4));
  dim3 gt(Dm / 32, Dm / 32);
  cvt_transpose_bf16<<<gt, 256, 0, stream>>>(Wq, Wqt, Dm, Dm);
  cvt_transpose_bf16<<<gt, 256, 0, stream>>>(Wk, Wkt, Dm, Dm);
  cvt_transpose_bf16<<<gt, 256, 0, stream>>>(Wv, Wvt, Dm, Dm);
  cvt_transpose_bf16<<<gt, 256, 0, stream>>>(Wo, Wot, Dm, Dm);

  // 2) QKV projections; 1/sqrt(DK)=0.125 folded into Q (exact, 2^-3)
  dim3 gg(Dm / 128, Mrows / 128);              // (8, 64)
  gemm_bf16<0><<<gg, 256, 0, stream>>>(Xb, Wqt, Qb,  Mrows, Dm, Dm, 0.125f);
  gemm_bf16<0><<<gg, 256, 0, stream>>>(Xb, Wkt, Kb,  Mrows, Dm, Dm, 1.0f);
  gemm_bf16<1><<<gg, 256, 0, stream>>>(Xb, Wvt, Vtb, Mrows, Dm, Dm, 1.0f);

  // 3) flash attention
  flash_attn<<<dim3(SeqN / 64, Bsz * Hh), 128, 0, stream>>>(Qb, Kb, Vtb, Oc);

  // 4) output projection -> f32
  gemm_bf16<2><<<gg, 256, 0, stream>>>(Oc, Wot, d_out, Mrows, Dm, Dm, 1.0f);
}
